// LayoutLMv3SelfAttention_40372692582559
// MI455X (gfx1250) — matrix operand order repaired and checked by reading, not compiled, not run on hardware
//
#include <hip/hip_runtime.h>
#include <hip/hip_bf16.h>

typedef __attribute__((ext_vector_type(16))) _Float16 v16h;
typedef __attribute__((ext_vector_type(8)))  _Float16 v8h;
typedef __attribute__((ext_vector_type(4)))  _Float16 v4h;
typedef __attribute__((ext_vector_type(8)))  float    v8f;

#define B_   2
#define S_   1024
#define HID_ 768
#define NH_  12
#define HD_  64
#define SCALE_ 0.125f   // 1/sqrt(64)

union AFrag { v16h v; v8h h[2]; };
union Pack8 { v8h v; _Float16 e[8]; };

static __device__ __forceinline__ void wait_dscnt0() {
#if __has_builtin(__builtin_amdgcn_s_wait_dscnt)
    __builtin_amdgcn_s_wait_dscnt(0);
#else
    asm volatile("s_wait_dscnt 0" ::: "memory");
#endif
}

// ---------------------------------------------------------------------------
// Kernel 1: fused QKV projection:  out[m,n] = sum_k hs[m,k]*W[n,k] + b[n]
// Block = 256 thr (8 waves). Block tile M=128 x N=64; wave tile 16x64.
// K staged 32 at a time (float4 global loads -> f16 LDS).
// Writes f16: Q (pre-scaled) [b,h,s,d], Kt [b,h,d,s] (packed b128 stores),
// V [b,h,s,d].
// ---------------------------------------------------------------------------
__global__ __launch_bounds__(256)
void qkv_proj_kernel(const float* __restrict__ hs,
                     const float* __restrict__ Wq, const float* __restrict__ bq,
                     const float* __restrict__ Wk, const float* __restrict__ bk,
                     const float* __restrict__ Wv, const float* __restrict__ bv,
                     _Float16* __restrict__ qbuf,
                     _Float16* __restrict__ ktbuf,
                     _Float16* __restrict__ vbuf)
{
    __shared__ _Float16 lA[128 * 32];   // [row][k]   8 KB
    __shared__ _Float16 lB[32 * 64];    // [k][n]     4 KB (W transposed)

    const int t    = threadIdx.x;
    const int lane = t & 31;
    const int wave = t >> 5;
    const int m0   = blockIdx.x * 128;
    const int n0   = blockIdx.y * 64;        // global out column in [0, 3*768)
    const int kind = n0 / HID_;              // 0=q 1=k 2=v
    const int c0   = n0 % HID_;

    const float* __restrict__ W    = (kind == 0) ? Wq : (kind == 1) ? Wk : Wv;
    const float* __restrict__ bias = (kind == 0) ? bq : (kind == 1) ? bk : bv;

    v8f acc[4] = {v8f{}, v8f{}, v8f{}, v8f{}};

    const int arow = wave * 16 + (lane & 15);
    const int kh0  = (lane < 16) ? 0 : 8;

#pragma unroll 1
    for (int k0 = 0; k0 < HID_; k0 += 32) {
        // stage A: 128x32 via float4 loads (1024 vec4), ds_store_b64
#pragma unroll
        for (int i = 0; i < 4; ++i) {
            int idx = t + i * 256;           // vec4 index
            int r = idx >> 3, kq = (idx & 7) * 4;
            const float4 f = *(const float4*)&hs[(m0 + r) * HID_ + k0 + kq];
            v4h hh; hh.x = (_Float16)f.x; hh.y = (_Float16)f.y;
            hh.z = (_Float16)f.z; hh.w = (_Float16)f.w;
            *(v4h*)&lA[r * 32 + kq] = hh;
        }
        // stage W transposed: lB[kk][n] = W[c0+n][k0+kk], float4 loads
#pragma unroll
        for (int i = 0; i < 2; ++i) {
            int idx = t + i * 256;           // 512 vec4 = 64 rows x 8 groups
            int n = idx >> 3, kq = (idx & 7) * 4;
            const float4 f = *(const float4*)&W[(c0 + n) * HID_ + k0 + kq];
            lB[(kq + 0) * 64 + n] = (_Float16)f.x;
            lB[(kq + 1) * 64 + n] = (_Float16)f.y;
            lB[(kq + 2) * 64 + n] = (_Float16)f.z;
            lB[(kq + 3) * 64 + n] = (_Float16)f.w;
        }
        __syncthreads();

        AFrag a;
        a.h[0] = *(const v8h*)&lA[arow * 32 + kh0];
        a.h[1] = *(const v8h*)&lA[arow * 32 + kh0 + 16];

#pragma unroll
        for (int nt = 0; nt < 4; ++nt) {
            AFrag b;
#pragma unroll
            for (int s = 0; s < 16; ++s)
                b.v[s] = lB[((s < 8) ? kh0 + s : kh0 + 8 + s) * 64 + nt * 16 + (lane & 15)];
            acc[nt] = __builtin_amdgcn_wmma_f32_16x16x32_f16(
                false, a.v, false, b.v, (short)0, acc[nt], false, false);
        }
        __syncthreads();
    }

    // Epilogue (kind is block-uniform -> single branch, no per-element selects).
    const int rowoff = (lane >> 4) << 3;
    const int lcol   = lane & 15;
    const int rbase  = m0 + wave * 16 + rowoff;      // 8 consecutive rows
    const int bb     = rbase >> 10;                  // batch (row block never straddles)
    const int s0v    = rbase & 1023;

    if (kind == 1) {
        // Kt[b,h,d,s]: 8 regs = 8 consecutive s at fixed d -> one b128 store
#pragma unroll
        for (int nt = 0; nt < 4; ++nt) {
            int c = c0 + nt * 16 + lcol, h = c >> 6, d = c & 63;
            float bval = bias[c];
            Pack8 pk;
#pragma unroll
            for (int r = 0; r < 8; ++r) pk.e[r] = (_Float16)(acc[nt][r] + bval);
            *(v8h*)&ktbuf[((bb * NH_ + h) * HD_ + d) * S_ + s0v] = pk.v;
        }
    } else if (kind == 0) {
#pragma unroll
        for (int nt = 0; nt < 4; ++nt) {
            int c = c0 + nt * 16 + lcol, h = c >> 6, d = c & 63;
            float bval = bias[c];
            _Float16* base = qbuf + ((bb * NH_ + h) * S_ + s0v) * HD_ + d;
#pragma unroll
            for (int r = 0; r < 8; ++r)
                base[r * HD_] = (_Float16)((acc[nt][r] + bval) * SCALE_);
        }
    } else {
#pragma unroll
        for (int nt = 0; nt < 4; ++nt) {
            int c = c0 + nt * 16 + lcol, h = c >> 6, d = c & 63;
            float bval = bias[c];
            _Float16* base = vbuf + ((bb * NH_ + h) * S_ + s0v) * HD_ + d;
#pragma unroll
            for (int r = 0; r < 8; ++r)
                base[r * HD_] = (_Float16)(acc[nt][r] + bval);
        }
    }
}

// ---------------------------------------------------------------------------
// Kernel 2: fused flash-attention with the 201 MB bias stream as wide b128
// loads. Block = 128 thr (4 waves); wave owns 16 query rows; 32-key chunks.
//   bias chunk (16x32 f32): float4 loads -> (rp+r2)*scale -> per-wave LDS
//   -> ds_load_b32 in WMMA C layout -> C operand of QK^T WMMA
//   online softmax (width-16 shfl_xor), P->A-frag via LDS, PV 4 WMMAs.
// ---------------------------------------------------------------------------
__global__ __launch_bounds__(128)
void attn_kernel(const _Float16* __restrict__ qbuf,
                 const _Float16* __restrict__ ktbuf,
                 const _Float16* __restrict__ vbuf,
                 const float* __restrict__ rel_pos,
                 const float* __restrict__ rel_2d,
                 const float* __restrict__ mask,
                 float* __restrict__ out)
{
    __shared__ float    lBias[4][16 * 32];  // per-wave bias tile (2 KB each)
    __shared__ _Float16 lP[4][16 * 32];     // per-wave P tile (1 KB each)

    const int t    = threadIdx.x;
    const int lane = t & 31;
    const int wave = t >> 5;
    const int bh   = blockIdx.x >> 4;
    const int qb   = blockIdx.x & 15;
    const int b    = bh / NH_, h = bh % NH_;
    const int q0   = qb * 64 + wave * 16;

    const _Float16* __restrict__ qp  = qbuf  + (bh * S_ + q0) * HD_;
    const _Float16* __restrict__ ktp = ktbuf + bh * HD_ * S_;
    const _Float16* __restrict__ vp  = vbuf  + bh * S_ * HD_;
    const float*    __restrict__ rp  = rel_pos + (size_t)bh * S_ * S_;
    const float*    __restrict__ r2  = rel_2d  + (size_t)bh * S_ * S_;

    // Q A-fragments for k-dim slices [0,32) and [32,64)
    const int arow = lane & 15;
    const int kh0  = (lane < 16) ? 0 : 8;
    AFrag qa0, qa1;
    qa0.h[0] = *(const v8h*)(qp + arow * HD_ + kh0);
    qa0.h[1] = *(const v8h*)(qp + arow * HD_ + kh0 + 16);
    qa1.h[0] = *(const v8h*)(qp + arow * HD_ + 32 + kh0);
    qa1.h[1] = *(const v8h*)(qp + arow * HD_ + 32 + kh0 + 16);

    const int rowoff = (lane >> 4) << 3;
    const int lcol   = lane & 15;

    float mrun[8], lrun[8];
#pragma unroll
    for (int r = 0; r < 8; ++r) { mrun[r] = -1e30f; lrun[r] = 0.f; }
    v8f o[4] = {v8f{}, v8f{}, v8f{}, v8f{}};

    float*    lb = lBias[wave];
    _Float16* lp = lP[wave];

    // bias staging geometry: 16 rows x 32 cols f32; lane handles one float4
    const int brow  = lane >> 3;            // +4 per iteration
    const int bcol4 = (lane & 7) * 4;

#pragma unroll 1
    for (int j0 = 0; j0 < S_; j0 += 32) {
        if (j0 + 32 < S_) {                 // global_prefetch_b8 lookahead
            __builtin_prefetch(rp + (q0 + brow) * S_ + j0 + 32 + bcol4, 0, 1);
            __builtin_prefetch(r2 + (q0 + brow) * S_ + j0 + 32 + bcol4, 0, 1);
        }

        // stage combined bias tile: (rp + r2) * scale, wide b128 loads
#pragma unroll
        for (int i = 0; i < 4; ++i) {
            const int row = brow + i * 4;
            const float4 a4 = *(const float4*)&rp[(q0 + row) * S_ + j0 + bcol4];
            const float4 b4 = *(const float4*)&r2[(q0 + row) * S_ + j0 + bcol4];
            float4 s4;
            s4.x = (a4.x + b4.x) * SCALE_;  s4.y = (a4.y + b4.y) * SCALE_;
            s4.z = (a4.z + b4.z) * SCALE_;  s4.w = (a4.w + b4.w) * SCALE_;
            *(float4*)&lb[row * 32 + bcol4] = s4;
        }
        wait_dscnt0();

        v8f sc[2];
#pragma unroll
        for (int kt = 0; kt < 2; ++kt) {
            const int key0 = j0 + kt * 16;
            const float mk = mask[b * S_ + key0 + lcol];

            // bias in WMMA C layout straight from LDS
            v8f c;
#pragma unroll
            for (int r = 0; r < 8; ++r)
                c[r] = lb[(r + rowoff) * 32 + kt * 16 + lcol] + mk;

            // Kt B-fragments: lane%16 = key, elements = d in table order
            AFrag kb0, kb1;
#pragma unroll
            for (int s = 0; s < 16; ++s) {
                const int kd = (s < 8) ? kh0 + s : kh0 + 8 + s;
                kb0.v[s] = ktp[(0  + kd) * S_ + key0 + lcol];
                kb1.v[s] = ktp[(32 + kd) * S_ + key0 + lcol];
            }

            v8f s = __builtin_amdgcn_wmma_f32_16x16x32_f16(
                false, qa0.v, false, kb0.v, (short)0, c, false, false);
            sc[kt] = __builtin_amdgcn_wmma_f32_16x16x32_f16(
                false, qa1.v, false, kb1.v, (short)0, s, false, false);
        }

        // online softmax over 16x32 chunk (row reductions across 16 lanes)
#pragma unroll
        for (int r = 0; r < 8; ++r) {
            float mx = fmaxf(sc[0][r], sc[1][r]);
            mx = fmaxf(mx, __shfl_xor(mx, 1, 16));
            mx = fmaxf(mx, __shfl_xor(mx, 2, 16));
            mx = fmaxf(mx, __shfl_xor(mx, 4, 16));
            mx = fmaxf(mx, __shfl_xor(mx, 8, 16));
            const float mnew = fmaxf(mrun[r], mx);
            const float corr = __expf(mrun[r] - mnew);
            mrun[r] = mnew;
            const float p0 = __expf(sc[0][r] - mnew);
            const float p1 = __expf(sc[1][r] - mnew);
            sc[0][r] = p0; sc[1][r] = p1;
            float rs = p0 + p1;
            rs += __shfl_xor(rs, 1, 16);
            rs += __shfl_xor(rs, 2, 16);
            rs += __shfl_xor(rs, 4, 16);
            rs += __shfl_xor(rs, 8, 16);
            lrun[r] = lrun[r] * corr + rs;
            o[0][r] *= corr; o[1][r] *= corr; o[2][r] *= corr; o[3][r] *= corr;
        }

        // P: C layout (f32) -> A-fragment layout (f16) via per-wave LDS tile
#pragma unroll
        for (int r = 0; r < 8; ++r) {
            lp[(r + rowoff) * 32 + lcol]      = (_Float16)sc[0][r];
            lp[(r + rowoff) * 32 + 16 + lcol] = (_Float16)sc[1][r];
        }
        wait_dscnt0();
        AFrag pa;
        pa.h[0] = *(const v8h*)&lp[arow * 32 + kh0];
        pa.h[1] = *(const v8h*)&lp[arow * 32 + kh0 + 16];

        // PV: V B-fragments (lane%16 = d, elements = key in table order)
#pragma unroll
        for (int nt = 0; nt < 4; ++nt) {
            AFrag vb;
#pragma unroll
            for (int s = 0; s < 16; ++s)
                vb.v[s] = vp[(j0 + ((s < 8) ? kh0 + s : kh0 + 8 + s)) * HD_ + nt * 16 + lcol];
            o[nt] = __builtin_amdgcn_wmma_f32_16x16x32_f16(
                false, pa.v, false, vb.v, (short)0, o[nt], false, false);
        }
        wait_dscnt0();   // WAR: this chunk's ds_loads before next chunk's stores
    }

    // Normalize and write ctx: [b, s, h*64 + d] fp32
#pragma unroll
    for (int r = 0; r < 8; ++r) {
        const float inv = 1.f / lrun[r];
        const int row = q0 + r + rowoff;
#pragma unroll
        for (int nt = 0; nt < 4; ++nt) {
            const int col = h * HD_ + nt * 16 + lcol;
            out[(b * S_ + row) * HID_ + col] = o[nt][r] * inv;
        }
    }
}

// ---------------------------------------------------------------------------
extern "C" void kernel_launch(void* const* d_in, const int* in_sizes, int n_in,
                              void* d_out, int out_size, void* d_ws, size_t ws_size,
                              hipStream_t stream) {
    const float* hs   = (const float*)d_in[0];
    const float* mask = (const float*)d_in[1];
    const float* rp   = (const float*)d_in[2];
    const float* r2   = (const float*)d_in[3];
    const float* Wq   = (const float*)d_in[4];
    const float* bq   = (const float*)d_in[5];
    const float* Wk   = (const float*)d_in[6];
    const float* bk   = (const float*)d_in[7];
    const float* Wv   = (const float*)d_in[8];
    const float* bv   = (const float*)d_in[9];
    float* out = (float*)d_out;

    const size_t nElem = (size_t)B_ * NH_ * S_ * HD_;   // 1.5M elems, 3 MB f16
    _Float16* qbuf  = (_Float16*)d_ws;
    _Float16* ktbuf = qbuf  + nElem;
    _Float16* vbuf  = ktbuf + nElem;

    dim3 g1(16, 36), b1(256);
    hipLaunchKernelGGL(qkv_proj_kernel, g1, b1, 0, stream,
                       hs, Wq, bq, Wk, bk, Wv, bv, qbuf, ktbuf, vbuf);

    dim3 g2(B_ * NH_ * (S_ / 64)), b2(128);
    hipLaunchKernelGGL(attn_kernel, g2, b2, 0, stream,
                       qbuf, ktbuf, vbuf, rp, r2, mask, out);
}
